// FrontToBackNetPart_52699248722026
// MI455X (gfx1250) — compile-verified, matrix operand order checked by reading, not run on hardware
//
#include <hip/hip_runtime.h>

// ---------------------------------------------------------------------------
// Types / constants
// ---------------------------------------------------------------------------
typedef __attribute__((ext_vector_type(16))) _Float16 v16h;
typedef __attribute__((ext_vector_type(8)))  _Float16 v8h;
typedef __attribute__((ext_vector_type(8)))  float    v8f;

union H16 { v16h v; v8h h[2]; };

#define BATCH   8
#define NPTS0   16384
#define SPTS1   512
#define SPTS2   128
#define NPARTS  14
#define BN_SC   0.9999950000374997f   // 1/sqrt(1+1e-5)

__host__ __device__ static inline int pad32i(int x) { return (x + 31) & ~31; }
__host__ __device__ static inline int pad64i(int x) { return (x + 63) & ~63; }

// ---------------------------------------------------------------------------
// Weight fold + swizzle into WMMA B-fragment order.
// Fragment layout (16-bit, 16x32 tile, per CDNA5 ISA 7.12.2):
//   lane = (n & 15) | (((k>>3)&1) << 4)
//   idx  = (k & 7)  | (((k>>4)&1) << 3)
// Tile storage: [ntile][ktile][lane 0..31][16 halves]
// npad is a multiple of 64 (the GEMM block-tile N) so the GEMM needs NO
// column guards: pad columns hold zero weights / zero bias.
// Also folds eval-BN:  W' = W * g * BN_SC ;  b' = b*g*BN_SC + be
// ---------------------------------------------------------------------------
__global__ void k_convert(const float* __restrict__ W, const float* __restrict__ b,
                          const float* __restrict__ g, const float* __restrict__ be,
                          int cin, int cout, int kpad, int npad,
                          _Float16* __restrict__ wz, float* __restrict__ bz)
{
    int t = blockIdx.x * 256 + threadIdx.x;
    if (t < npad) {
        float v = 0.f;
        if (t < cout) {
            float s = g ? g[t] * BN_SC : 1.f;
            v = b[t] * s + (be ? be[t] : 0.f);
        }
        bz[t] = v;
    }
    if (t >= kpad * npad) return;
    int k = t / npad;
    int n = t - k * npad;
    float val = 0.f;
    if (k < cin && n < cout) {
        float s = g ? g[n] * BN_SC : 1.f;
        val = W[(size_t)k * cout + n] * s;
    }
    int lane = (n & 15) | (((k >> 3) & 1) << 4);
    int idx  = (k & 7)  | (((k >> 4) & 1) << 3);
    size_t pos = ((size_t)(n >> 4) * (kpad >> 5) + (k >> 5)) * 512 + lane * 16 + idx;
    wz[pos] = (_Float16)val;
}

// ---------------------------------------------------------------------------
// Generic fused GEMM:  out = act( A[M x Kpad] * Wswz[Kpad x Npad64] + bias )
//   A: fp16, row stride = strideA (multiple of 32 halves, rows 64B aligned)
//   Block: 256 threads = 8 waves; block tile 128(M) x 64(N);
//   wave -> 32x32 via 2x2 WMMA 16x16x32 f16->f32 tiles.
//   A tiles are software-pipelined: global b128 loads for tile kt+1 are
//   issued right after tile kt's LDS store, overlapping HBM latency with WMMA.
//   Weights are pre-swizzled + padded to 64 cols -> branch-free B loads.
//   STORE_H : write fp16 activation (stride strideO, pads zero-filled)
//   STORE_F : write fp32 [M x nout]
//   FUSE_Z  : off-head grouped conv: z[row, col/128] += relu(val) * gw[col]
// ---------------------------------------------------------------------------
template<bool RELU, bool STORE_H, bool STORE_F, bool FUSE_Z>
__global__ void __launch_bounds__(256)
k_gemm(const _Float16* __restrict__ A, int strideA,
       const _Float16* __restrict__ Wz, int kpad,
       const float* __restrict__ bias,
       _Float16* __restrict__ outH, int strideO,
       float* __restrict__ outF, int nout,
       const float* __restrict__ gw, float* __restrict__ zbuf)
{
    const int KT  = kpad >> 5;
    const int m0  = blockIdx.x * 128;
    const int n0  = blockIdx.y * 64;
    const int tid = threadIdx.x;
    const int lane = tid & 31;
    const int wv   = tid >> 5;
    const int mw   = wv >> 1;     // 0..3 : which 32-row band
    const int nw   = wv & 1;      // 0..1 : which 32-col band

    __shared__ __align__(16) _Float16 lsA[8 * 32 * 16];   // 128x32 tile, fragment-swizzled

    __builtin_prefetch((const void*)Wz, 0, 0);            // global_prefetch_b8 (L2 warm)

    v8f acc[2][2];
#pragma unroll
    for (int i = 0; i < 2; ++i)
#pragma unroll
        for (int j = 0; j < 2; ++j)
#pragma unroll
            for (int e = 0; e < 8; ++e) acc[i][j][e] = 0.f;

    // ---- per-thread A-staging geometry (2 chunks of 8 halves each)
    int rowi[2], k0i[2], ldsOff[2];
    const _Float16* aSrc[2];
#pragma unroll
    for (int i = 0; i < 2; ++i) {
        int chunk = tid * 2 + i;              // 512 chunks cover 128x32 halves
        rowi[i]   = chunk >> 2;               // 0..127
        k0i[i]    = (chunk & 3) * 8;          // 0,8,16,24
        ldsOff[i] = (((rowi[i] >> 4) * 32) + ((rowi[i] & 15) | ((k0i[i] & 8) ? 16 : 0))) * 16
                    + ((k0i[i] & 16) ? 8 : 0);
        aSrc[i]   = A + (size_t)(m0 + rowi[i]) * strideA + k0i[i];
    }
    // prologue fetch of tile 0
    v8h gbuf[2];
#pragma unroll
    for (int i = 0; i < 2; ++i) gbuf[i] = *(const v8h*)aSrc[i];

    for (int kt = 0; kt < KT; ++kt) {
        __syncthreads();                      // prev compute done reading lsA
        // ---- commit current tile to LDS (pre-swizzled, ds_store_b128)
#pragma unroll
        for (int i = 0; i < 2; ++i) *(v8h*)&lsA[ldsOff[i]] = gbuf[i];
        // ---- issue next tile's global loads (branchless; overlaps WMMA below)
        int ktn = (kt + 1 < KT) ? (kt + 1) : kt;
#pragma unroll
        for (int i = 0; i < 2; ++i) gbuf[i] = *(const v8h*)(aSrc[i] + (size_t)ktn * 32);
        __syncthreads();

        // ---- B fragments straight from global (padded to 64 cols: no guard)
        H16 bfr[2];
#pragma unroll
        for (int ni = 0; ni < 2; ++ni) {
            int gn = n0 + nw * 32 + ni * 16;
            const _Float16* p = Wz + (((size_t)(gn >> 4) * KT + kt) * 32 + lane) * 16;
            bfr[ni].h[0] = *(const v8h*)p;
            bfr[ni].h[1] = *(const v8h*)(p + 8);
        }
        // ---- A fragments from LDS (2x ds_load_b128 each)
        H16 afr[2];
#pragma unroll
        for (int ti = 0; ti < 2; ++ti) {
            const _Float16* p = &lsA[((mw * 2 + ti) * 32 + lane) * 16];
            afr[ti].h[0] = *(const v8h*)p;
            afr[ti].h[1] = *(const v8h*)(p + 8);
        }
        // ---- 4x WMMA
#pragma unroll
        for (int ti = 0; ti < 2; ++ti)
#pragma unroll
            for (int ni = 0; ni < 2; ++ni)
                acc[ti][ni] = __builtin_amdgcn_wmma_f32_16x16x32_f16(
                    false, afr[ti].v, false, bfr[ni].v,
                    (short)0, acc[ti][ni], false, false);
    }

    // ---- epilogue: bias + activation + stores (C layout per ISA 7.12.2)
#pragma unroll
    for (int ti = 0; ti < 2; ++ti) {
#pragma unroll
        for (int ni = 0; ni < 2; ++ni) {
            int colB = n0 + nw * 32 + ni * 16 + (lane & 15);
            float bv = bias[colB];            // bias padded to 64-col multiple
            int rbase = m0 + (mw * 2 + ti) * 16 + ((lane >> 4) << 3);
#pragma unroll
            for (int v = 0; v < 8; ++v) {
                float val = acc[ti][ni][v] + bv;
                if (RELU) val = fmaxf(val, 0.f);
                int row = rbase + v;
                if (STORE_H) {
                    if (colB < strideO)
                        outH[(size_t)row * strideO + colB] = (_Float16)val;
                }
                if (STORE_F) {
                    if (colB < nout)
                        outF[(size_t)row * nout + colB] = val;
                }
                if (FUSE_Z) {
                    if (colB < nout)
                        atomicAdd(&zbuf[(size_t)row * NPARTS + (colB >> 7)], val * gw[colB]);
                }
            }
        }
    }
}

// ---------------------------------------------------------------------------
// Farthest point sampling (one block per batch; dist in global ws; LDS argmax)
// ---------------------------------------------------------------------------
__global__ void k_fps(const float* __restrict__ xyz, int Npt, int S,
                      float* __restrict__ dist, int* __restrict__ idx,
                      float* __restrict__ newxyz)
{
    int b = blockIdx.x, tid = threadIdx.x;
    __shared__ float sv[256];
    __shared__ int   si[256];
    __shared__ int   sFar;
    const float* X = xyz + (size_t)b * Npt * 3;
    float* D = dist + (size_t)b * Npt;
    for (int j = tid; j < Npt; j += 256) D[j] = 1e10f;
    if (tid == 0) sFar = 0;
    __syncthreads();
    for (int it = 0; it < S; ++it) {
        int far = sFar;
        if (tid == 0) idx[b * S + it] = far;
        if (tid < 3) newxyz[((size_t)b * S + it) * 3 + tid] = X[(size_t)far * 3 + tid];
        float cx = X[far * 3 + 0], cy = X[far * 3 + 1], cz = X[far * 3 + 2];
        float bm = -1.f; int bi = 0;
        for (int j = tid; j < Npt; j += 256) {
            float dx = X[j * 3 + 0] - cx, dy = X[j * 3 + 1] - cy, dz = X[j * 3 + 2] - cz;
            float d = dx * dx + dy * dy + dz * dz;
            float dd = fminf(D[j], d);
            D[j] = dd;
            if (dd > bm) { bm = dd; bi = j; }
        }
        sv[tid] = bm; si[tid] = bi;
        __syncthreads();
        for (int st = 128; st > 0; st >>= 1) {
            if (tid < st && sv[tid + st] > sv[tid]) { sv[tid] = sv[tid + st]; si[tid] = si[tid + st]; }
            __syncthreads();
        }
        if (tid == 0) sFar = si[0];
        __syncthreads();
    }
}

// ---------------------------------------------------------------------------
// Ball query: first K (ascending index) neighbors within radius; fill w/ first
// ---------------------------------------------------------------------------
__global__ void k_ballq(const float* __restrict__ xyz, int Npt,
                        const float* __restrict__ newxyz, int S,
                        float r2, int K, int* __restrict__ idxo, int total)
{
    int t = blockIdx.x * 256 + threadIdx.x;
    if (t >= total) return;
    int b = t / S;
    const float* X = xyz + (size_t)b * Npt * 3;
    const float* q = newxyz + (size_t)t * 3;
    int* o = idxo + (size_t)t * K;
    float qx = q[0], qy = q[1], qz = q[2];
    int cnt = 0;
    for (int j = 0; j < Npt && cnt < K; ++j) {
        float dx = X[j * 3 + 0] - qx, dy = X[j * 3 + 1] - qy, dz = X[j * 3 + 2] - qz;
        if (dx * dx + dy * dy + dz * dz <= r2) o[cnt++] = j;
    }
    int f = (cnt > 0) ? o[0] : 0;
    for (; cnt < K; ++cnt) o[cnt] = f;
}

// ---------------------------------------------------------------------------
// Pack grouped features: [feats(idx) , xyz(idx)-new_xyz] -> fp16 padded matrix
// ---------------------------------------------------------------------------
__global__ void k_pack_group(const float* __restrict__ xyz, const float* __restrict__ newxyz,
                             const int* __restrict__ idx,
                             const _Float16* __restrict__ featsH, int featStride, int C,
                             int S, int K, int Npt, int strideO,
                             _Float16* __restrict__ out, int rows)
{
    int r = blockIdx.x * 256 + threadIdx.x;
    if (r >= rows) return;
    int bs = r / K;
    int b  = bs / S;
    int id = idx[r];
    const float* q = newxyz + (size_t)bs * 3;
    const float* p = xyz + ((size_t)b * Npt + id) * 3;
    _Float16* o = out + (size_t)r * strideO;
    int c = 0;
    if (featsH) {
        const _Float16* f = featsH + ((size_t)b * Npt + id) * featStride;
        for (; c < C; ++c) o[c] = f[c];
    } else {
        for (; c < C; ++c) o[c] = (_Float16)p[c];     // C==3: points are xyz
    }
    o[C + 0] = (_Float16)(p[0] - q[0]);
    o[C + 1] = (_Float16)(p[1] - q[1]);
    o[C + 2] = (_Float16)(p[2] - q[2]);
    for (c = C + 3; c < strideO; ++c) o[c] = (_Float16)0.f;
}

// max over K group members, write into concat feature buffer at column offset
__global__ void k_maxpool(const _Float16* __restrict__ act, int stride, int K, int C,
                          int coff, int catStride, _Float16* __restrict__ out, int total)
{
    int t = blockIdx.x * 256 + threadIdx.x;
    if (t >= total) return;
    int c = t % C, bs = t / C;
    const _Float16* p = act + (size_t)bs * K * stride + c;
    float m = -1e30f;
    for (int k = 0; k < K; ++k) m = fmaxf(m, (float)p[(size_t)k * stride]);
    out[(size_t)bs * catStride + coff + c] = (_Float16)m;
}

// sa3 group-all input:  [l2_xyz(3) , l2_feats(512)] -> 544-stride fp16
__global__ void k_pack_sa3(const float* __restrict__ xyz2, const _Float16* __restrict__ feats,
                           _Float16* __restrict__ out, int rows)
{
    int r = blockIdx.x * 256 + threadIdx.x;
    if (r >= rows) return;
    _Float16* o = out + (size_t)r * 544;
    const float* q = xyz2 + (size_t)r * 3;
    o[0] = (_Float16)q[0]; o[1] = (_Float16)q[1]; o[2] = (_Float16)q[2];
    const _Float16* f = feats + (size_t)r * 512;
    for (int c = 0; c < 512; ++c) o[3 + c] = f[c];
    for (int c = 515; c < 544; ++c) o[c] = (_Float16)0.f;
}

// global max over 128 points per batch (sa3 output, stride 1024)
__global__ void k_maxpool_l3(const _Float16* __restrict__ act, _Float16* __restrict__ l3, int total)
{
    int t = blockIdx.x * 256 + threadIdx.x;
    if (t >= total) return;
    int c = t % 1024, b = t / 1024;
    const _Float16* p = act + (size_t)b * 128 * 1024 + c;
    float m = -1e30f;
    for (int k = 0; k < 128; ++k) m = fmaxf(m, (float)p[(size_t)k * 1024]);
    l3[t] = (_Float16)m;
}

// fp3 input: [l2_feats(512) , broadcast l3(1024)] -> 1536-stride fp16
__global__ void k_pack_fp3(const _Float16* __restrict__ feats, const _Float16* __restrict__ l3,
                           _Float16* __restrict__ out, int rows)
{
    int r = blockIdx.x * 256 + threadIdx.x;
    if (r >= rows) return;
    int b = r / 128;
    _Float16* o = out + (size_t)r * 1536;
    const _Float16* f = feats + (size_t)r * 512;
    for (int c = 0; c < 512; ++c) o[c] = f[c];
    const _Float16* g = l3 + (size_t)b * 1024;
    for (int c = 0; c < 1024; ++c) o[512 + c] = g[c];
}

// ---------------------------------------------------------------------------
// FP layers: 3-NN inverse-distance interpolation + concat -> fp16 padded input
//   dupXyz!=0 : points1 = [xyz,xyz] (6 channels of raw coords)
// ---------------------------------------------------------------------------
__global__ void k_fp_pack(const float* __restrict__ xyz1, int N1,
                          const float* __restrict__ xyz2, int N2,
                          const _Float16* __restrict__ p1H, int s1, int C1, int dupXyz,
                          const _Float16* __restrict__ p2H, int s2, int C2,
                          int strideO, _Float16* __restrict__ out, int rows)
{
    int r = blockIdx.x * 256 + threadIdx.x;
    if (r >= rows) return;
    int b = r / N1;
    const float* q = xyz1 + (size_t)r * 3;
    float qx = q[0], qy = q[1], qz = q[2];
    const float* X2 = xyz2 + (size_t)b * N2 * 3;
    float d0 = 1e30f, d1 = 1e30f, d2 = 1e30f;
    int i0 = 0, i1 = 0, i2 = 0;
    for (int j = 0; j < N2; ++j) {
        float dx = X2[j * 3 + 0] - qx, dy = X2[j * 3 + 1] - qy, dz = X2[j * 3 + 2] - qz;
        float d = dx * dx + dy * dy + dz * dz;
        if (d < d0)      { d2 = d1; i2 = i1; d1 = d0; i1 = i0; d0 = d; i0 = j; }
        else if (d < d1) { d2 = d1; i2 = i1; d1 = d;  i1 = j; }
        else if (d < d2) { d2 = d;  i2 = j; }
    }
    float w0 = 1.f / (d0 + 1e-8f), w1 = 1.f / (d1 + 1e-8f), w2 = 1.f / (d2 + 1e-8f);
    float ws = w0 + w1 + w2;
    w0 /= ws; w1 /= ws; w2 /= ws;
    _Float16* o = out + (size_t)r * strideO;
    if (dupXyz) {
        for (int c = 0; c < C1; ++c) o[c] = (_Float16)q[c % 3];
    } else {
        const _Float16* f = p1H + (size_t)r * s1;
        for (int c = 0; c < C1; ++c) o[c] = f[c];
    }
    const _Float16* a0 = p2H + ((size_t)b * N2 + i0) * s2;
    const _Float16* a1 = p2H + ((size_t)b * N2 + i1) * s2;
    const _Float16* a2 = p2H + ((size_t)b * N2 + i2) * s2;
    for (int c = 0; c < C2; ++c)
        o[C1 + c] = (_Float16)(w0 * (float)a0[c] + w1 * (float)a1[c] + w2 * (float)a2[c]);
    for (int c = C1 + C2; c < strideO; ++c) o[c] = (_Float16)0.f;
}

__global__ void k_zero_f(float* __restrict__ p, long long n)
{
    long long t = (long long)blockIdx.x * 256 + threadIdx.x;
    if (t < n) p[t] = 0.f;
}

// ---------------------------------------------------------------------------
// Final head: softmax(logits) ; z = zbuf + off_b ; dz = sum(z*p) ; back coords
// ---------------------------------------------------------------------------
__global__ void k_final(const float* __restrict__ logits, const float* __restrict__ zbuf,
                        const float* __restrict__ offb, const float* __restrict__ xyz,
                        float* __restrict__ back, int total)
{
    int t = blockIdx.x * 256 + threadIdx.x;
    if (t >= total) return;
    const float* L = logits + (size_t)t * NPARTS;
    float mx = -1e30f;
    for (int k = 0; k < NPARTS; ++k) mx = fmaxf(mx, L[k]);
    float se = 0.f, ez[NPARTS];
    for (int k = 0; k < NPARTS; ++k) { ez[k] = __expf(L[k] - mx); se += ez[k]; }
    float inv = 1.f / se, dz = 0.f;
    for (int k = 0; k < NPARTS; ++k)
        dz += (zbuf[(size_t)t * NPARTS + k] + offb[k]) * ez[k] * inv;
    back[(size_t)t * 3 + 0] = xyz[(size_t)t * 3 + 0];
    back[(size_t)t * 3 + 1] = xyz[(size_t)t * 3 + 1];
    back[(size_t)t * 3 + 2] = xyz[(size_t)t * 3 + 2] + dz;
}

// ---------------------------------------------------------------------------
// Host orchestration
// ---------------------------------------------------------------------------
struct Layer { const float *W, *b, *g, *be; int cin, cout; _Float16* wz; float* bz; };

extern "C" void kernel_launch(void* const* d_in, const int* in_sizes, int n_in,
                              void* d_out, int out_size, void* d_ws, size_t ws_size,
                              hipStream_t stream)
{
    (void)in_sizes; (void)n_in; (void)out_size; (void)ws_size;
    const float* xyz = (const float*)d_in[0];
    int pi = 1;
    auto nextp = [&]() { return (const float*)d_in[pi++]; };
    auto mk = [&](int cin, int cout) {
        Layer L{}; L.W = nextp(); L.b = nextp(); L.g = nextp(); L.be = nextp();
        L.cin = cin; L.cout = cout; return L;
    };
    Layer sa1_0[3] = { mk(6, 32),   mk(32, 32),   mk(32, 64) };
    Layer sa1_1[3] = { mk(6, 64),   mk(64, 64),   mk(64, 128) };
    Layer sa1_2[3] = { mk(6, 64),   mk(64, 96),   mk(96, 128) };
    Layer sa2_0[3] = { mk(323, 128), mk(128, 128), mk(128, 256) };
    Layer sa2_1[3] = { mk(323, 128), mk(128, 196), mk(196, 256) };
    Layer sa3l[3]  = { mk(515, 256), mk(256, 512), mk(512, 1024) };
    Layer fp3l[2]  = { mk(1536, 256), mk(256, 256) };
    Layer fp2l[2]  = { mk(576, 256),  mk(256, 128) };
    Layer fp1l[2]  = { mk(134, 128),  mk(128, 128) };
    Layer seml[1]  = { mk(128, 128) };
    Layer semo{}; semo.W = nextp(); semo.b = nextp(); semo.g = nullptr; semo.be = nullptr;
    semo.cin = 128; semo.cout = NPARTS;
    Layer offl[1]  = { mk(128, 128 * NPARTS) };
    const float* offOutW = nextp();   // [14,128] flat
    const float* offOutB = nextp();   // [14]

    // ---- deterministic bump allocator over d_ws
    char* wsBase = (char*)d_ws;
    size_t wsOff = 0;
    auto walloc = [&](size_t bytes) -> void* {
        void* p = wsBase + wsOff;
        wsOff = (wsOff + bytes + 255) & ~(size_t)255;
        return p;
    };

    // ---- fold+swizzle all weights (cheap, re-done every launch: deterministic)
    auto conv = [&](Layer& L) {
        int kp = pad32i(L.cin), np = pad64i(L.cout);   // N padded to block tile
        L.wz = (_Float16*)walloc((size_t)kp * np * 2);
        L.bz = (float*)walloc((size_t)np * 4);
        int tot = kp * np;
        k_convert<<<dim3((tot + 255) / 256), dim3(256), 0, stream>>>(
            L.W, L.b, L.g, L.be, L.cin, L.cout, kp, np, L.wz, L.bz);
    };
    for (int i = 0; i < 3; ++i) conv(sa1_0[i]);
    for (int i = 0; i < 3; ++i) conv(sa1_1[i]);
    for (int i = 0; i < 3; ++i) conv(sa1_2[i]);
    for (int i = 0; i < 3; ++i) conv(sa2_0[i]);
    for (int i = 0; i < 3; ++i) conv(sa2_1[i]);
    for (int i = 0; i < 3; ++i) conv(sa3l[i]);
    for (int i = 0; i < 2; ++i) conv(fp3l[i]);
    for (int i = 0; i < 2; ++i) conv(fp2l[i]);
    for (int i = 0; i < 2; ++i) conv(fp1l[i]);
    conv(seml[0]); conv(semo); conv(offl[0]);

    // ---- scratch buffers
    const size_t MAXROWS = (size_t)BATCH * SPTS1 * 128;            // 524288
    _Float16* bufA   = (_Float16*)walloc(MAXROWS * 160 * 2);       // widest pack stride
    _Float16* bufB   = (_Float16*)walloc(MAXROWS * 128 * 2);
    float*    fpsD   = (float*)walloc((size_t)BATCH * NPTS0 * 4);
    int*      idx1   = (int*)walloc((size_t)BATCH * SPTS1 * 4);
    float*    new1   = (float*)walloc((size_t)BATCH * SPTS1 * 3 * 4);
    int*      idx2   = (int*)walloc((size_t)BATCH * SPTS2 * 4);
    float*    new2   = (float*)walloc((size_t)BATCH * SPTS2 * 3 * 4);
    int*      ballIx = (int*)walloc(MAXROWS * 4);
    _Float16* l1f    = (_Float16*)walloc((size_t)BATCH * SPTS1 * 320 * 2);
    _Float16* l2f    = (_Float16*)walloc((size_t)BATCH * SPTS2 * 512 * 2);
    _Float16* l3h    = (_Float16*)walloc((size_t)BATCH * 1024 * 2);
    _Float16* fp3o   = (_Float16*)walloc((size_t)BATCH * SPTS2 * 256 * 2);
    _Float16* fp2o   = (_Float16*)walloc((size_t)BATCH * SPTS1 * 128 * 2);
    _Float16* l0feat = (_Float16*)walloc((size_t)BATCH * NPTS0 * 128 * 2);
    float*    zbuf   = (float*)walloc((size_t)BATCH * NPTS0 * NPARTS * 4);

    float* outLogits = (float*)d_out;
    float* outBack   = outLogits + (size_t)BATCH * NPTS0 * NPARTS;

    // ---- fused GEMM+bias+ReLU -> fp16 activations
    auto gemmH = [&](const _Float16* A, Layer& L, _Float16* outH, int M) {
        int kp = pad32i(L.cin), so = pad32i(L.cout);
        dim3 g(M / 128, (so + 63) / 64);
        k_gemm<true, true, false, false><<<g, dim3(256), 0, stream>>>(
            A, kp, L.wz, kp, L.bz, outH, so, nullptr, 0, nullptr, nullptr);
    };

    // ================= SA1 (MSG over 16384 -> 512 centroids) =================
    k_fps<<<dim3(BATCH), dim3(256), 0, stream>>>(xyz, NPTS0, SPTS1, fpsD, idx1, new1);
    {
        const float r1[3] = { 0.1f, 0.2f, 0.4f };
        const int   k1[3] = { 32, 64, 128 };
        const int   co[3] = { 0, 64, 192 };
        Layer* Ls[3] = { sa1_0, sa1_1, sa1_2 };
        for (int br = 0; br < 3; ++br) {
            int K = k1[br], rows = BATCH * SPTS1 * K, qs = BATCH * SPTS1;
            k_ballq<<<dim3((qs + 255) / 256), dim3(256), 0, stream>>>(
                xyz, NPTS0, new1, SPTS1, r1[br] * r1[br], K, ballIx, qs);
            k_pack_group<<<dim3((rows + 255) / 256), dim3(256), 0, stream>>>(
                xyz, new1, ballIx, nullptr, 0, 3, SPTS1, K, NPTS0, 32, bufA, rows);
            gemmH(bufA, Ls[br][0], bufB, rows);
            gemmH(bufB, Ls[br][1], bufA, rows);
            gemmH(bufA, Ls[br][2], bufB, rows);
            int C = Ls[br][2].cout, tot = BATCH * SPTS1 * C;
            k_maxpool<<<dim3((tot + 255) / 256), dim3(256), 0, stream>>>(
                bufB, pad32i(C), K, C, co[br], 320, l1f, tot);
        }
    }
    // ================= SA2 (MSG over 512 -> 128 centroids) ===================
    k_fps<<<dim3(BATCH), dim3(256), 0, stream>>>(new1, SPTS1, SPTS2, fpsD, idx2, new2);
    {
        const float r2a[2] = { 0.4f, 0.8f };
        const int   k2[2]  = { 64, 128 };
        const int   co[2]  = { 0, 256 };
        Layer* Ls[2] = { sa2_0, sa2_1 };
        for (int br = 0; br < 2; ++br) {
            int K = k2[br], rows = BATCH * SPTS2 * K, qs = BATCH * SPTS2;
            k_ballq<<<dim3((qs + 255) / 256), dim3(256), 0, stream>>>(
                new1, SPTS1, new2, SPTS2, r2a[br] * r2a[br], K, ballIx, qs);
            k_pack_group<<<dim3((rows + 255) / 256), dim3(256), 0, stream>>>(
                new1, new2, ballIx, l1f, 320, 320, SPTS2, K, SPTS1, 352, bufA, rows);
            gemmH(bufA, Ls[br][0], bufB, rows);
            gemmH(bufB, Ls[br][1], bufA, rows);
            gemmH(bufA, Ls[br][2], bufB, rows);
            int C = Ls[br][2].cout, tot = BATCH * SPTS2 * C;
            k_maxpool<<<dim3((tot + 255) / 256), dim3(256), 0, stream>>>(
                bufB, pad32i(C), K, C, co[br], 512, l2f, tot);
        }
    }
    // ================= SA3 (group-all, 128 pts -> global 1024) ===============
    {
        int rows = BATCH * SPTS2;
        k_pack_sa3<<<dim3((rows + 255) / 256), dim3(256), 0, stream>>>(new2, l2f, bufA, rows);
        gemmH(bufA, sa3l[0], bufB, rows);
        gemmH(bufB, sa3l[1], bufA, rows);
        gemmH(bufA, sa3l[2], bufB, rows);
        int tot = BATCH * 1024;
        k_maxpool_l3<<<dim3((tot + 255) / 256), dim3(256), 0, stream>>>(bufB, l3h, tot);
    }
    // ================= FP3 / FP2 / FP1 =======================================
    {
        int rows = BATCH * SPTS2;
        k_pack_fp3<<<dim3((rows + 255) / 256), dim3(256), 0, stream>>>(l2f, l3h, bufA, rows);
        gemmH(bufA, fp3l[0], bufB, rows);
        gemmH(bufB, fp3l[1], fp3o, rows);
    }
    {
        int rows = BATCH * SPTS1;
        k_fp_pack<<<dim3((rows + 255) / 256), dim3(256), 0, stream>>>(
            new1, SPTS1, new2, SPTS2, l1f, 320, 320, 0, fp3o, 256, 256, 576, bufA, rows);
        gemmH(bufA, fp2l[0], bufB, rows);
        gemmH(bufB, fp2l[1], fp2o, rows);
    }
    {
        int rows = BATCH * NPTS0;
        k_fp_pack<<<dim3((rows + 255) / 256), dim3(256), 0, stream>>>(
            xyz, NPTS0, new1, SPTS1, nullptr, 0, 6, 1, fp2o, 128, 128, 160, bufA, rows);
        gemmH(bufA, fp1l[0], bufB, rows);
        gemmH(bufB, fp1l[1], l0feat, rows);
    }
    // ================= heads ================================================
    int M = BATCH * NPTS0;
    gemmH(l0feat, seml[0], bufA, M);                       // sem hidden (128, ReLU)
    {   // logits = hidden @ semW + b  (no BN/ReLU), f32 straight into d_out
        dim3 g(M / 128, 1);
        k_gemm<false, false, true, false><<<g, dim3(256), 0, stream>>>(
            bufA, 128, semo.wz, 128, semo.bz,
            nullptr, 0, outLogits, NPARTS, nullptr, nullptr);
    }
    {   // offset head fused: relu(l0feat @ Woff' + boff') grouped-dot into zbuf
        long long zn = (long long)M * NPARTS;
        k_zero_f<<<dim3((unsigned)((zn + 255) / 256)), dim3(256), 0, stream>>>(zbuf, zn);
        dim3 g(M / 128, (1792 + 63) / 64);
        k_gemm<true, false, false, true><<<g, dim3(256), 0, stream>>>(
            l0feat, 128, offl[0].wz, 128, offl[0].bz,
            nullptr, 0, nullptr, 1792, offOutW, zbuf);
    }
    k_final<<<dim3((M + 255) / 256), dim3(256), 0, stream>>>(
        outLogits, zbuf, offOutB, xyz, outBack, M);
}
